// DisConv_20650202759324
// MI455X (gfx1250) — matrix operand order repaired and checked by reading, not compiled, not run on hardware
//
#include <hip/hip_runtime.h>
#include <hip/hip_bf16.h>

// Problem constants (match reference)
#define Nn      2048
#define IN_DIMn 128
#define KCH     8
#define DD      32
#define ITERSn  4
#define EPSn    1e-12f

typedef __attribute__((ext_vector_type(16))) _Float16 v16h;
typedef __attribute__((ext_vector_type(8)))  float    v8f;

union F16x16 { v16h v; _Float16 e[16]; unsigned u[8]; };

// Low 32 bits of a generic pointer to __shared__ == wave-relative LDS byte
// offset (flat-aperture mapping, ISA 10.2), which is what async dsaddr wants.
__device__ __forceinline__ unsigned lds_off(const void* p) {
    return (unsigned)(unsigned long long)p;
}

// CDNA5 async memory->LDS copies (ASYNCcnt, no VGPR round-trip). GVS mode:
// global address is a 64-bit SGPR base plus a 32-bit VGPR offset; LDS address comes from the VDST VGPR.
__device__ __forceinline__ void async_ld_lds_b128(unsigned lds, unsigned goff,
                                                  const void* base) {
    asm volatile("global_load_async_to_lds_b128 %0, %1, %2"
                 :: "v"(lds), "v"(goff), "s"((unsigned long long)base)
                 : "memory");
}
__device__ __forceinline__ void async_ld_lds_b64(unsigned lds, unsigned goff,
                                                 const void* base) {
    asm volatile("global_load_async_to_lds_b64 %0, %1, %2"
                 :: "v"(lds), "v"(goff), "s"((unsigned long long)base)
                 : "memory");
}
__device__ __forceinline__ void wait_async0() {
    asm volatile("s_wait_asynccnt 0x0" ::: "memory");
}

// ---------------------------------------------------------------------------
// Projection: Z[k,n,:] = l2norm(features[n,:] @ W[k] + b[k])
// One wave per (k,n); lane = output dim c (D == wave width == 32).
// ---------------------------------------------------------------------------
__global__ __launch_bounds__(256) void disconv_proj(
    const float* __restrict__ feat, const float* __restrict__ W,
    const float* __restrict__ b, float* __restrict__ Z,
    _Float16* __restrict__ Zh)
{
    int gw   = (blockIdx.x * blockDim.x + threadIdx.x) >> 5; // global wave id
    int lane = threadIdx.x & 31;
    int k = gw >> 11;          // gw / N   (N = 2048)
    int n = gw & (Nn - 1);

    float acc = b[k * DD + lane];
    const float* frow = feat + (size_t)n * IN_DIMn;
    const float* wcol = W + (size_t)k * IN_DIMn * DD + lane;
    #pragma unroll 4
    for (int i = 0; i < IN_DIMn; ++i)
        acc = fmaf(frow[i], wcol[(size_t)i * DD], acc);

    float s = acc * acc;
    #pragma unroll
    for (int m = 16; m >= 1; m >>= 1) s += __shfl_xor(s, m, 32);
    float scale = 1.0f / fmaxf(sqrtf(s), EPSn);
    float zv = acc * scale;

    size_t idx = ((size_t)k * Nn + n) * DD + lane;
    Z[idx]  = zv;
    Zh[idx] = (_Float16)zv;
}

// ---------------------------------------------------------------------------
// One routing iteration, fully fused:
//   att = Z Z^T (per channel, WMMA f16->f32), channel-axis softmax with adj
//   mask, Z' = l2norm(Z + att Z)   (aggregate via WMMA)
// Block = 16 output rows x all 8 channels. Wave k owns channel k.
// ---------------------------------------------------------------------------
__global__ __launch_bounds__(256) void disconv_route(
    const float* __restrict__ Zin, const _Float16* __restrict__ Zhin,
    const int* __restrict__ adj,
    float* __restrict__ Zout, _Float16* __restrict__ Zhout,
    float* __restrict__ out, int write_out)
{
    __shared__ __align__(16) _Float16 sZh[KCH][32][DD]; // m-tile of Z (16KB)
    __shared__ __align__(16) float sExp[KCH][16][32];   // masked exp    (16KB)
    __shared__ __align__(16) float sRcp[16][32];        // 1/denominator (2KB)
    __shared__ __align__(16) int   sAdj[16][32];        // adjacency     (2KB)

    const int t    = threadIdx.x;
    const int lane = t & 31;
    const int k    = t >> 5;        // channel owned by this wave
    const int h    = lane >> 4;     // half-wave
    const int nl   = lane & 15;
    const int n0   = blockIdx.x * 16;

    // A fragment (16x32 f16, M=n rows, K=D): Z[k, n0+nl, :]
    F16x16 Au;
    #pragma unroll
    for (int v = 0; v < 8; ++v) {
        int d0 = 16 * h + 2 * v;
        Au.u[v] = *(const unsigned*)(Zhin + ((size_t)k * Nn + n0 + nl) * DD + d0);
    }

    // Accumulators in C/D layout (row = v + 8h, col = nl [+16]); init = residual Z
    v8f acc0, acc1;
    #pragma unroll
    for (int v = 0; v < 8; ++v) {
        int row = v + 8 * h;
        acc0[v] = Zin[((size_t)k * Nn + n0 + row) * DD + nl];
        acc1[v] = Zin[((size_t)k * Nn + n0 + row) * DD + 16 + nl];
    }

    // Async-staging geometry (per thread, per m-tile):
    //   Zh tile: 1024 x 16B chunks; chunk = kk*128 + mm*4 + part
    //   adj tile: 256 x 8B chunks;  chunk t -> row t>>4, cols (t&15)*2
    const unsigned zhLds  = lds_off(&sZh[0][0][0]);
    const unsigned adjLds = lds_off(&sAdj[0][0]);

    for (int m0 = 0; m0 < Nn; m0 += 32) {
        __syncthreads();   // previous tile's LDS consumers are done

        // Stage Zh m-tile (all channels) + adj tile via async global->LDS DMA
        #pragma unroll
        for (int j = 0; j < 4; ++j) {
            int chunk = t + 256 * j;
            int kk   = chunk >> 7;
            int mm   = (chunk >> 2) & 31;
            int part = chunk & 3;
            unsigned goff = (unsigned)(kk * Nn + m0 + mm) * (DD * 2) + part * 16;
            async_ld_lds_b128(zhLds + (unsigned)chunk * 16, goff, Zhin);
        }
        {
            unsigned goff = (unsigned)((n0 + (t >> 4)) * Nn + m0 + (t & 15) * 2) * 4;
            async_ld_lds_b64(adjLds + (unsigned)t * 8, goff, adj);
        }
        if (m0 + 32 < Nn)   // speculative prefetch of next adj tile
            __builtin_prefetch(adj + (size_t)(n0 + (t >> 4)) * Nn + m0 + 32, 0, 1);
        wait_async0();
        __syncthreads();

        // Attention scores for this channel: two 16x16 WMMAs, masked exp -> LDS
        #pragma unroll
        for (int msub = 0; msub < 2; ++msub) {
            F16x16 Bu;   // B fragment (32x16, K=D, N=m): B[d,m] = Zh[m0+msub*16+m][d]
            #pragma unroll
            for (int v = 0; v < 8; ++v)
                Bu.u[v] = *(const unsigned*)&sZh[k][msub * 16 + nl][16 * h + 2 * v];
            v8f c0 = {};
            v8f att = __builtin_amdgcn_wmma_f32_16x16x32_f16(
                false, Au.v, false, Bu.v, (short)0, c0, false, false);
            #pragma unroll
            for (int v = 0; v < 8; ++v) {
                int row = v + 8 * h;
                int mc  = msub * 16 + nl;
                sExp[k][row][mc] = (sAdj[row][mc] > 0) ? __expf(att[v]) : 0.0f;
            }
        }
        __syncthreads();

        // Channel-axis softmax: store reciprocal denominators (mask is k-indep.)
        for (int e = t; e < 512; e += 256) {
            int nn = e >> 5, mm = e & 31;
            float s = 0.0f;
            #pragma unroll
            for (int kk = 0; kk < KCH; ++kk) s += sExp[kk][nn][mm];
            sRcp[nn][mm] = (s > 0.0f) ? __builtin_amdgcn_rcpf(s) : 0.0f;
        }
        __syncthreads();

        // P (16x32 probs) as f16 A-fragment (M=n, K=m), built straight from LDS
        F16x16 Pu;
        #pragma unroll
        for (int v = 0; v < 8; ++v) {
            int mp = 16 * h + 2 * v;
            Pu.e[2 * v]     = (_Float16)(sExp[k][nl][mp]     * sRcp[nl][mp]);
            Pu.e[2 * v + 1] = (_Float16)(sExp[k][nl][mp + 1] * sRcp[nl][mp + 1]);
        }

        // Aggregate: acc += P (16x32) x Zh_tile (32x32), two 16x16 output blocks
        #pragma unroll
        for (int csub = 0; csub < 2; ++csub) {
            F16x16 Bu;   // B fragment (32x16, K=m, N=c): B[m,c] = Zh[m][csub*16+c]
            #pragma unroll
            for (int v = 0; v < 8; ++v) {
                Bu.e[2 * v]     = sZh[k][16 * h + 2 * v    ][csub * 16 + nl];
                Bu.e[2 * v + 1] = sZh[k][16 * h + 2 * v + 1][csub * 16 + nl];
            }
            if (csub == 0)
                acc0 = __builtin_amdgcn_wmma_f32_16x16x32_f16(
                    false, Pu.v, false, Bu.v, (short)0, acc0, false, false);
            else
                acc1 = __builtin_amdgcn_wmma_f32_16x16x32_f16(
                    false, Pu.v, false, Bu.v, (short)0, acc1, false, false);
        }
    }

    // Row L2 norm: each row lives in one VGPR index across a 16-lane half
    #pragma unroll
    for (int v = 0; v < 8; ++v) {
        float s = acc0[v] * acc0[v] + acc1[v] * acc1[v];
        s += __shfl_xor(s, 1, 32);
        s += __shfl_xor(s, 2, 32);
        s += __shfl_xor(s, 4, 32);
        s += __shfl_xor(s, 8, 32);
        float scale = 1.0f / fmaxf(sqrtf(s), EPSn);
        acc0[v] *= scale;
        acc1[v] *= scale;
    }

    #pragma unroll
    for (int v = 0; v < 8; ++v) {
        int row = v + 8 * h;
        size_t i0 = ((size_t)k * Nn + n0 + row) * DD + nl;
        Zout[i0]       = acc0[v];
        Zout[i0 + 16]  = acc1[v];
        Zhout[i0]      = (_Float16)acc0[v];
        Zhout[i0 + 16] = (_Float16)acc1[v];
        if (write_out) {
            out[(size_t)(n0 + row) * (KCH * DD) + k * DD + nl]      = acc0[v];
            out[(size_t)(n0 + row) * (KCH * DD) + k * DD + 16 + nl] = acc1[v];
        }
    }
}

// ---------------------------------------------------------------------------
extern "C" void kernel_launch(void* const* d_in, const int* in_sizes, int n_in,
                              void* d_out, int out_size, void* d_ws, size_t ws_size,
                              hipStream_t stream)
{
    const int*   adj  = (const int*)d_in[0];
    const float* feat = (const float*)d_in[1];
    const float* W    = (const float*)d_in[2];
    const float* b    = (const float*)d_in[3];
    float* out = (float*)d_out;

    const size_t ZE = (size_t)KCH * Nn * DD;     // elements per Z buffer
    float*    Z0  = (float*)d_ws;
    float*    Z1  = Z0 + ZE;
    _Float16* Zh0 = (_Float16*)(Z1 + ZE);
    _Float16* Zh1 = Zh0 + ZE;                    // total ws use: 6 MB

    disconv_proj<<<(KCH * Nn * 32) / 256, 256, 0, stream>>>(feat, W, b, Z0, Zh0);

    float* zi = Z0; _Float16* zhi = Zh0;
    float* zo = Z1; _Float16* zho = Zh1;
    for (int it = 0; it < ITERSn; ++it) {
        disconv_route<<<Nn / 16, 256, 0, stream>>>(zi, zhi, adj, zo, zho, out,
                                                   it == ITERSn - 1);
        float* tz = zi; zi = zo; zo = tz;
        _Float16* th = zhi; zhi = zho; zho = th;
    }
    (void)in_sizes; (void)n_in; (void)out_size; (void)ws_size;
}
